// CausalSelfAttention_27951647162988
// MI455X (gfx1250) — matrix operand order repaired and checked by reading, not compiled, not run on hardware
//
#include <hip/hip_runtime.h>
#include <hip/hip_bf16.h>

typedef __attribute__((ext_vector_type(16))) _Float16 v16h;
typedef __attribute__((ext_vector_type(8)))  _Float16 v8h;
typedef __attribute__((ext_vector_type(8)))  float    v8f;

// Problem constants
#define BATCH   2
#define TSEQ    2048
#define NEMBD   768
#define NHEAD   12
#define HDIM    64
#define BH      (BATCH * NHEAD)     // 24
#define MROWS   (BATCH * TSEQ)      // 4096
#define NQKV    (3 * NEMBD)         // 2304
#define TPAD    2080                // TSEQ + 32 pad for V^T tail loads

__device__ __forceinline__ v8f wmma_f16(v16h a, v16h b, v8f c) {
    // D = A(16x32,f16) * B(32x16,f16) + C(16x16,f32)
    return __builtin_amdgcn_wmma_f32_16x16x32_f16(
        false, a, false, b, (short)0, c, false, false);
}

// A-fragment (16x32 f16): lane holds row M = lane&15; element e -> K = lh*16+e
// (lh = lane>>4).  Two contiguous 16B chunks.
__device__ __forceinline__ v16h load_a_frag(const _Float16* row_k0, int lh) {
    const _Float16* p = row_k0 + lh * 16;
    v8h lo = *(const v8h*)(p);
    v8h hi = *(const v8h*)(p + 8);
    v16h a;
#pragma unroll
    for (int i = 0; i < 8; ++i) { a[i] = lo[i]; a[i + 8] = hi[i]; }
    return a;
}

// ---------------- conversion kernels ----------------
__global__ void cvt_x_kernel(const float* __restrict__ x, _Float16* __restrict__ xh, int n) {
    int i = blockIdx.x * blockDim.x + threadIdx.x;
    if (i < n) xh[i] = (_Float16)x[i];
}

// wT[n*K + k] = w[k*N + n]  (f32 -> f16 transpose so B-fragments load contiguously)
__global__ void cvt_wT_kernel(const float* __restrict__ w, _Float16* __restrict__ wT, int K, int N) {
    int i = blockIdx.x * blockDim.x + threadIdx.x;
    if (i < K * N) {
        int k = i / N, n = i % N;
        wT[(size_t)n * K + k] = (_Float16)w[i];
    }
}

__global__ void zero_vt_pad_kernel(_Float16* __restrict__ Vt) {
    int i = blockIdx.x * blockDim.x + threadIdx.x;   // 24*64*32 pad elems
    if (i < BH * HDIM * (TPAD - TSEQ)) {
        int row = i >> 5;                 // bh*64 + d
        int t   = TSEQ + (i & 31);
        Vt[(size_t)row * TPAD + t] = (_Float16)0.f;
    }
}

// ---------------- QKV GEMM: [4096,768] @ [768,2304] ----------------
// One wave computes a 32x64 tile. Scatters into Q (scaled), K, V^T layouts.
__global__ void __launch_bounds__(128) qkv_gemm_kernel(
        const _Float16* __restrict__ xh, const _Float16* __restrict__ wT,
        _Float16* __restrict__ Qh, _Float16* __restrict__ Kh, _Float16* __restrict__ Vt) {
    int lane = threadIdx.x & 31;
    int wid  = (blockIdx.x * blockDim.x + threadIdx.x) >> 5;
    const int NT = NQKV / 64;            // 36
    const int MT = MROWS / 32;           // 128
    if (wid >= MT * NT) return;
    int nt = wid % NT, mt = wid / NT;
    int m0 = mt * 32, n0 = nt * 64;
    int lm = lane & 15, lh = lane >> 4;

    v8f acc[2][4] = {};
    const _Float16* arow0 = xh + (size_t)(m0 + lm) * NEMBD;
    const _Float16* arow1 = xh + (size_t)(m0 + 16 + lm) * NEMBD;

    for (int k0 = 0; k0 < NEMBD; k0 += 32) {
        v16h a0 = load_a_frag(arow0 + k0, lh);
        v16h a1 = load_a_frag(arow1 + k0, lh);
#pragma unroll
        for (int j = 0; j < 4; ++j) {
            const _Float16* bp = wT + (size_t)(n0 + j * 16 + lm) * NEMBD + k0 + lh * 16;
            v16h b = *(const v16h*)bp;
            acc[0][j] = wmma_f16(a0, b, acc[0][j]);
            acc[1][j] = wmma_f16(a1, b, acc[1][j]);
        }
    }
#pragma unroll
    for (int i = 0; i < 2; ++i)
#pragma unroll
        for (int j = 0; j < 4; ++j)
#pragma unroll
            for (int v = 0; v < 8; ++v) {
                int m = m0 + i * 16 + v + 8 * lh;
                int n = n0 + j * 16 + lm;
                float val = acc[i][j][v];
                int b = m >> 11, t = m & (TSEQ - 1);
                if (n < NEMBD) {                       // Q, pre-scaled by 1/sqrt(64)
                    int h = n >> 6, d = n & 63;
                    Qh[(((size_t)(b * NHEAD + h)) * TSEQ + t) * HDIM + d] = (_Float16)(val * 0.125f);
                } else if (n < 2 * NEMBD) {            // K
                    int c = n - NEMBD; int h = c >> 6, d = c & 63;
                    Kh[(((size_t)(b * NHEAD + h)) * TSEQ + t) * HDIM + d] = (_Float16)val;
                } else {                               // V transposed: [bh][d][t]
                    int c = n - 2 * NEMBD; int h = c >> 6, d = c & 63;
                    Vt[(((size_t)(b * NHEAD + h)) * HDIM + d) * TPAD + t] = (_Float16)val;
                }
            }
}

// ---------------- Flash attention: one wave per (bh, 16-query tile) ----------------
#define ATT_WAVES 4
__global__ void __launch_bounds__(128) attn_kernel(
        const _Float16* __restrict__ Qh, const _Float16* __restrict__ Kh,
        const _Float16* __restrict__ Vt, _Float16* __restrict__ Oh) {
    __shared__ __align__(32) _Float16 Pbuf[ATT_WAVES][16 * 32];
    int lane  = threadIdx.x & 31;
    int wslot = threadIdx.x >> 5;
    int wid   = blockIdx.x * ATT_WAVES + wslot;
    if (wid >= BH * (TSEQ / 16)) return;
    int bh = wid >> 7;                  // /128 query tiles
    int qt = wid & 127;
    int qbase = qt * 16;
    int lm = lane & 15, lh = lane >> 4;

    const _Float16* Qp = Qh + ((size_t)bh * TSEQ + qbase + lm) * HDIM;
    v16h qlo = load_a_frag(Qp, lh);         // d 0..31
    v16h qhi = load_a_frag(Qp + 32, lh);    // d 32..63

    const _Float16* Kbase = Kh + (size_t)bh * TSEQ * HDIM;
    const _Float16* Vbase = Vt + (size_t)bh * HDIM * TPAD;
    _Float16* P = &Pbuf[wslot][0];

    v8f acc[4] = {};
    float rmax[8], rsum[8];
#pragma unroll
    for (int v = 0; v < 8; ++v) { rmax[v] = -3.0e38f; rsum[v] = 0.f; }

    int kend = qbase + 16;
    for (int kt = 0; kt < kend; kt += 32) {
        float s[2][8];
#pragma unroll
        for (int sub = 0; sub < 2; ++sub) {
            int key  = kt + sub * 16 + lm;              // B-frag lane = key column
            int krow = key < TSEQ ? key : (TSEQ - 1);   // clamp OOB loads; masked anyway
            const _Float16* Kp = Kbase + (size_t)krow * HDIM + lh * 16;
            v16h blo = *(const v16h*)Kp;
            v16h bhi = *(const v16h*)(Kp + 32);
            v8f c = {};
            c = wmma_f16(qlo, blo, c);
            c = wmma_f16(qhi, bhi, c);
#pragma unroll
            for (int v = 0; v < 8; ++v) {
                int q = qbase + v + 8 * lh;
                s[sub][v] = (key > q) ? -1.0e30f : c[v];   // causal mask (Q pre-scaled)
            }
        }
        // online softmax: row stats via cross-lane reduce over the 16-lane N group
#pragma unroll
        for (int v = 0; v < 8; ++v) {
            float t = fmaxf(s[0][v], s[1][v]);
            t = fmaxf(t, __shfl_xor(t, 1, 32));
            t = fmaxf(t, __shfl_xor(t, 2, 32));
            t = fmaxf(t, __shfl_xor(t, 4, 32));
            t = fmaxf(t, __shfl_xor(t, 8, 32));
            float nm = fmaxf(rmax[v], t);
            float alpha = __expf(rmax[v] - nm);
            rmax[v] = nm;
            float p0 = __expf(s[0][v] - nm);
            float p1 = __expf(s[1][v] - nm);
            float ps = p0 + p1;
            ps += __shfl_xor(ps, 1, 32);
            ps += __shfl_xor(ps, 2, 32);
            ps += __shfl_xor(ps, 4, 32);
            ps += __shfl_xor(ps, 8, 32);
            rsum[v] = rsum[v] * alpha + ps;
#pragma unroll
            for (int j = 0; j < 4; ++j) acc[j][v] *= alpha;
            int m = v + 8 * lh;
            P[m * 32 + lm]      = (_Float16)p0;   // C-layout -> [m][k] tile in LDS
            P[m * 32 + 16 + lm] = (_Float16)p1;
        }
        asm volatile("s_wait_dscnt 0" ::: "memory");   // cross-lane LDS RAW within wave
        v16h pa = load_a_frag(P + lm * 32, lh);        // A-frag of P (16x32)
#pragma unroll
        for (int j = 0; j < 4; ++j) {
            const _Float16* Vp = Vbase + (size_t)(j * 16 + lm) * TPAD + kt + lh * 16;
            v16h vb = *(const v16h*)Vp;                // B-frag: lane = d col, elems = keys
            acc[j] = wmma_f16(pa, vb, acc[j]);
        }
    }

#pragma unroll
    for (int v = 0; v < 8; ++v) rsum[v] = 1.0f / rsum[v];
    int b = bh / NHEAD, h = bh % NHEAD;
#pragma unroll
    for (int j = 0; j < 4; ++j)
#pragma unroll
        for (int v = 0; v < 8; ++v) {
            int m   = qbase + v + 8 * lh;
            int col = h * HDIM + j * 16 + lm;
            Oh[((size_t)(b * TSEQ + m)) * NEMBD + col] = (_Float16)(acc[j][v] * rsum[v]);
        }
}

// ---------------- Proj GEMM: [4096,768] @ [768,768] -> fp32 out ----------------
__global__ void __launch_bounds__(128) proj_gemm_kernel(
        const _Float16* __restrict__ Oh, const _Float16* __restrict__ wT,
        float* __restrict__ out) {
    int lane = threadIdx.x & 31;
    int wid  = (blockIdx.x * blockDim.x + threadIdx.x) >> 5;
    const int NT = NEMBD / 64;           // 12
    const int MT = MROWS / 32;           // 128
    if (wid >= MT * NT) return;
    int nt = wid % NT, mt = wid / NT;
    int m0 = mt * 32, n0 = nt * 64;
    int lm = lane & 15, lh = lane >> 4;

    v8f acc[2][4] = {};
    const _Float16* arow0 = Oh + (size_t)(m0 + lm) * NEMBD;
    const _Float16* arow1 = Oh + (size_t)(m0 + 16 + lm) * NEMBD;

    for (int k0 = 0; k0 < NEMBD; k0 += 32) {
        v16h a0 = load_a_frag(arow0 + k0, lh);
        v16h a1 = load_a_frag(arow1 + k0, lh);
#pragma unroll
        for (int j = 0; j < 4; ++j) {
            const _Float16* bp = wT + (size_t)(n0 + j * 16 + lm) * NEMBD + k0 + lh * 16;
            v16h b = *(const v16h*)bp;
            acc[0][j] = wmma_f16(a0, b, acc[0][j]);
            acc[1][j] = wmma_f16(a1, b, acc[1][j]);
        }
    }
#pragma unroll
    for (int i = 0; i < 2; ++i)
#pragma unroll
        for (int j = 0; j < 4; ++j)
#pragma unroll
            for (int v = 0; v < 8; ++v) {
                int m = m0 + i * 16 + v + 8 * lh;
                int n = n0 + j * 16 + lm;
                out[(size_t)m * NEMBD + n] = acc[i][j][v];
            }
}

extern "C" void kernel_launch(void* const* d_in, const int* in_sizes, int n_in,
                              void* d_out, int out_size, void* d_ws, size_t ws_size,
                              hipStream_t stream) {
    const float* x      = (const float*)d_in[0];
    const float* w_qkv  = (const float*)d_in[1];
    const float* w_proj = (const float*)d_in[2];
    float* out = (float*)d_out;

    char* ws = (char*)d_ws;
    size_t off = 0;
    _Float16* xh     = (_Float16*)(ws + off); off += (size_t)MROWS * NEMBD * 2;
    _Float16* wqkvT  = (_Float16*)(ws + off); off += (size_t)NQKV  * NEMBD * 2;
    _Float16* wprojT = (_Float16*)(ws + off); off += (size_t)NEMBD * NEMBD * 2;
    _Float16* Qh     = (_Float16*)(ws + off); off += (size_t)BH * TSEQ * HDIM * 2;
    _Float16* Kh     = (_Float16*)(ws + off); off += (size_t)BH * TSEQ * HDIM * 2;
    _Float16* Vt     = (_Float16*)(ws + off); off += (size_t)BH * HDIM * TPAD * 2;
    _Float16* Oh     = (_Float16*)(ws + off); off += (size_t)MROWS * NEMBD * 2;

    int nx = MROWS * NEMBD;                                  // 3,145,728
    cvt_x_kernel<<<(nx + 255) / 256, 256, 0, stream>>>(x, xh, nx);
    int nwq = NEMBD * NQKV;                                  // 1,769,472
    cvt_wT_kernel<<<(nwq + 255) / 256, 256, 0, stream>>>(w_qkv, wqkvT, NEMBD, NQKV);
    int nwp = NEMBD * NEMBD;                                 // 589,824
    cvt_wT_kernel<<<(nwp + 255) / 256, 256, 0, stream>>>(w_proj, wprojT, NEMBD, NEMBD);
    int npad = BH * HDIM * (TPAD - TSEQ);                    // 49,152
    zero_vt_pad_kernel<<<(npad + 255) / 256, 256, 0, stream>>>(Vt);

    // QKV: 128 x 36 = 4608 waves, 4 waves/block
    qkv_gemm_kernel<<<4608 / 4, 128, 0, stream>>>(xh, wqkvT, Qh, Kh, Vt);
    // Attention: 24 * 128 = 3072 waves
    attn_kernel<<<3072 / ATT_WAVES, ATT_WAVES * 32, 0, stream>>>(Qh, Kh, Vt, Oh);
    // Proj: 128 x 12 = 1536 waves
    proj_gemm_kernel<<<1536 / 4, 128, 0, stream>>>(Oh, wprojT, out);
}
